// Sparse3DNA_79937931313216
// MI455X (gfx1250) — matrix operand order repaired and checked by reading, not compiled, not run on hardware
//
#include <hip/hip_runtime.h>
#include <hip/hip_bf16.h>

typedef __attribute__((ext_vector_type(16))) __bf16 v16bf;
typedef __attribute__((ext_vector_type(8)))  float  v8f;
typedef unsigned short ushort_t;
typedef unsigned int   uint_t;
typedef __attribute__((address_space(3))) ushort_t lds_ushort;

// ---------------------------------------------------------------------------
// Problem constants (from reference)
// ---------------------------------------------------------------------------
#define N_TOK   2048          // F*H*W = 2*32*32
#define INNER   512           // HEADS * DIM_HEAD
#define QKVW    1536          // 3 * INNER
#define HEADS   8
#define DHEAD   64
#define NWIN    126           // 1 BOS + 125 neighbors
#define MPAD    2064          // 2049 rows (tokens + zero pad) rounded up to 16

// ---------------------------------------------------------------------------
// s_wait_asynccnt 0 (builtin when available, else inline asm per ISA SOPP 74)
// ---------------------------------------------------------------------------
__device__ __forceinline__ void wait_async0() {
#if __has_builtin(__builtin_amdgcn_s_wait_asynccnt)
    __builtin_amdgcn_s_wait_asynccnt(0);
#else
    asm volatile("s_wait_asynccnt 0x0" ::: "memory");
#endif
}

// ---------------------------------------------------------------------------
// float -> bf16 (round to nearest even, ignoring NaN edge)
// ---------------------------------------------------------------------------
__device__ __forceinline__ ushort_t f2bf(float f) {
    uint_t u = __float_as_uint(f);
    uint_t r = u + 0x7FFFu + ((u >> 16) & 1u);
    return (ushort_t)(r >> 16);
}

// ---------------------------------------------------------------------------
// Elementwise convert (with zero tail padding): dst[i] = bf16(i < n_src ? src[i] : 0)
// ---------------------------------------------------------------------------
__global__ void cvt_pad_kernel(const float* __restrict__ src, ushort_t* __restrict__ dst,
                               int n_src, int n_dst) {
    int idx = blockIdx.x * blockDim.x + threadIdx.x;
    if (idx >= n_dst) return;
    float v = (idx < n_src) ? src[idx] : 0.0f;
    dst[idx] = f2bf(v);
}

// ---------------------------------------------------------------------------
// BOS row of the attention output: row 0 = V-section of qkv row 0
// ---------------------------------------------------------------------------
__global__ void bos_row_kernel(const float* __restrict__ qkv, ushort_t* __restrict__ ab) {
    int c = blockIdx.x * blockDim.x + threadIdx.x;
    if (c >= INNER) return;
    ab[c] = f2bf(qkv[2 * INNER + c]);   // qkv[0, 1024 + c]
}

// ---------------------------------------------------------------------------
// bf16 GEMM with f32 accumulation via v_wmma_f32_16x16x32_bf16.
//   C[M x N] = A[M x K] * B[K x N] (+ bias[N])
// Block = 128 threads = 4 waves; each wave computes one 16x16 tile, the block
// covers 16(M) x 64(N). The shared 32x64 bf16 B tile is staged into LDS with
// GLOBAL_LOAD_ASYNC_TO_LDS_B128 (ASYNCcnt) once per k-step and consumed by all
// 4 waves via 16-bit LDS reads. No divergence: EXEC is all-ones for every WMMA.
// ---------------------------------------------------------------------------
__global__ __launch_bounds__(128)
void gemm_bf16_kernel(const ushort_t* __restrict__ A, const ushort_t* __restrict__ B,
                      float* __restrict__ C, const float* __restrict__ bias,
                      int lda, int ldb, int ldc, int K) {
    __shared__ ushort_t s_b[32 * 64];     // 32(K) x 64(N) bf16 tile, 4 KB

    const int tid  = threadIdx.x;
    const int lane = tid & 31;
    const int wv   = tid >> 5;
    const int m0   = blockIdx.x * 16;
    const int bn0  = blockIdx.y * 64;     // block's N base
    const int n0   = bn0 + wv * 16;       // wave's N base

    // LDS byte offset of the staging buffer (addrspace(3) ptrtoint -> offset)
    const unsigned sbase = (unsigned)(unsigned long long)(lds_ushort*)s_b;

    // A fragment addressing (16-bit A 16x32, ISA 7.12.2):
    //   lane L: M = L%16 ; K-halves: lanes 0-15 -> K 0..7 & 16..23, lanes 16-31 -> +8
    const int arow = m0 + (lane & 15);
    const int akb  = (lane >> 4) << 3;            // 0 or 8
    const ushort_t* abase = A + (size_t)arow * lda + akb;

    // This thread's two 16-byte chunks of the B tile (256 chunks total):
    //   chunk c -> tile row c>>3, byte-in-row (c&7)*16  (8 ushorts)
    const int c0 = tid, c1 = tid + 128;

    v8f acc = {};
    for (int k0 = 0; k0 < K; k0 += 32) {
        // ---- stage B tile (32 x 64) into LDS asynchronously ----
        {
            const ushort_t* g0 = B + (size_t)(k0 + (c0 >> 3)) * ldb + bn0 + ((c0 & 7) << 3);
            const ushort_t* g1 = B + (size_t)(k0 + (c1 >> 3)) * ldb + bn0 + ((c1 & 7) << 3);
            asm volatile("global_load_async_to_lds_b128 %0, %1, off"
                         :: "v"(sbase + c0 * 16), "v"(g0) : "memory");
            asm volatile("global_load_async_to_lds_b128 %0, %1, off"
                         :: "v"(sbase + c1 * 16), "v"(g1) : "memory");
        }

        // ---- A fragment from global (overlaps with async B) ----
        union { uint4 u[2]; v16bf v; } af, bfm;
        af.u[0] = *(const uint4*)(abase + k0);        // K kb+0..7 (pairs per VGPR)
        af.u[1] = *(const uint4*)(abase + k0 + 16);   // K kb+16..23
        if (k0 + 32 < K)
            __builtin_prefetch((const void*)(abase + k0 + 32), 0, 3);

        wait_async0();        // own async writes landed in LDS
        __syncthreads();      // ... and everyone else's

        // ---- B fragment from LDS: lane L holds column N=L%16, K in table order (kb+0..7, kb+16..23) ----
        for (int s = 0; s < 16; ++s)
            ((ushort_t*)&bfm)[s] = s_b[(((lane >> 4) << 3) + (s & 7) + ((s >> 3) << 4)) * 64 + wv * 16 + (lane & 15)];

        acc = __builtin_amdgcn_wmma_f32_16x16x32_bf16(
            /*neg_a=*/false, af.v, /*neg_b=*/false, bfm.v,
            /*c_mod=*/(short)0, acc, /*reuse_a=*/false, /*reuse_b=*/false);

        __syncthreads();      // protect tile before next iteration overwrites it
    }

    // D layout: VGPR r -> M = m0 + 8*(lane>=16) + r, N = n0 + lane%16
    const int mrow = m0 + ((lane >> 4) << 3);
    const int col  = n0 + (lane & 15);
    const float bv = bias ? bias[col] : 0.0f;
#pragma unroll
    for (int r = 0; r < 8; ++r) {
        C[(size_t)(mrow + r) * ldc + col] = acc[r] + bv;
    }
}

// ---------------------------------------------------------------------------
// Windowed causal attention. One wave per (head, query).
// Phase 1: lanes parallel over window entries (4 each), full 64-d dot vs q.
// Softmax via wave32 butterfly reductions. Phase 2: lanes parallel over dims
// (2 each), sequential over the 126 window entries, weighted V accumulate.
// Output written as bf16 into attention-output matrix row i+1 (row 2048 dropped).
// ---------------------------------------------------------------------------
__global__ __launch_bounds__(128)
void attn_kernel(const float* __restrict__ qkv, ushort_t* __restrict__ ab) {
    __shared__ float s_q[4][64];
    __shared__ float s_p[4][128];
    __shared__ int   s_r[4][128];

    const int lane = threadIdx.x & 31;
    const int wv   = threadIdx.x >> 5;
    const int wid  = blockIdx.x * 4 + wv;     // 0 .. 16383
    const int h    = wid >> 11;               // head
    const int i    = wid & 2047;              // query grid index
    const int fz   = i >> 10;                 // frame (0..1)
    const int hy   = (i >> 5) & 31;
    const int wx   = i & 31;

    // q = qkv[i+1, h*64 + d] * scale  (scale = 1/sqrt(64) = 0.125)
    {
        const float* qp = qkv + (size_t)(i + 1) * QKVW + h * DHEAD;
        s_q[wv][lane]      = qp[lane]      * 0.125f;
        s_q[wv][lane + 32] = qp[lane + 32] * 0.125f;
    }
    __syncthreads();

    // ---- Phase 1: scores ----
    float sc[4];
    int   rw[4];
#pragma unroll
    for (int t = 0; t < 4; ++t) {
        const int j = lane + 32 * t;
        float s = -3.4e38f;
        int krow = 0;
        if (j < NWIN) {
            bool masked = false;
            if (j > 0) {
                const int jj = j - 1;                  // 0..124, df-major order
                const int nf = fz + (jj / 25) - 2;
                const int nh = hy + ((jj / 5) % 5) - 2;
                const int nw = wx + (jj % 5) - 2;
                int m;
                if (nf >= 0 && nf < 2 && nh >= 0 && nh < 32 && nw >= 0 && nw < 32)
                    m = (nf << 10) + (nh << 5) + nw;
                else
                    m = N_TOK;                         // OOB -> always masked
                masked = (i < m);
                krow = m + 1;                          // rows >= 2048 are zero
            }
            if (!masked) {
                const float* kp = qkv + (size_t)krow * QKVW + INNER + h * DHEAD;
                float acc = 0.0f;
#pragma unroll
                for (int dd = 0; dd < 16; ++dd) {
                    float4 kv = *(const float4*)(kp + dd * 4);
                    acc += s_q[wv][4 * dd + 0] * kv.x + s_q[wv][4 * dd + 1] * kv.y
                         + s_q[wv][4 * dd + 2] * kv.z + s_q[wv][4 * dd + 3] * kv.w;
                }
                s = acc;
            }
        }
        sc[t] = s;
        rw[t] = krow;
    }

    // ---- softmax (wave32 butterfly) ----
    float mx = fmaxf(fmaxf(sc[0], sc[1]), fmaxf(sc[2], sc[3]));
#pragma unroll
    for (int off = 16; off > 0; off >>= 1)
        mx = fmaxf(mx, __shfl_xor(mx, off, 32));

    float p[4];
    float sum = 0.0f;
#pragma unroll
    for (int t = 0; t < 4; ++t) {
        const int j = lane + 32 * t;
        p[t] = (j < NWIN) ? __expf(sc[t] - mx) : 0.0f;
        sum += p[t];
    }
#pragma unroll
    for (int off = 16; off > 0; off >>= 1)
        sum += __shfl_xor(sum, off, 32);
    const float inv = 1.0f / sum;   // BOS entry guarantees sum > 0

#pragma unroll
    for (int t = 0; t < 4; ++t) {
        const int j = lane + 32 * t;
        if (j < NWIN) { s_p[wv][j] = p[t]; s_r[wv][j] = rw[t]; }
    }
    __syncthreads();

    // ---- Phase 2: out[d] = sum_j p_j * V[row_j][d], lanes over d ----
    float a0 = 0.0f, a1 = 0.0f;
    const float* vb = qkv + 2 * INNER + h * DHEAD + 2 * lane;
    for (int j = 0; j < NWIN; ++j) {
        const int   r  = s_r[wv][j];
        const float pj = s_p[wv][j];
        float2 v = *(const float2*)(vb + (size_t)r * QKVW);
        a0 += pj * v.x;
        a1 += pj * v.y;
    }
    a0 *= inv;
    a1 *= inv;

    if (i + 1 < N_TOK) {   // query 2047's output row is dropped by the reference
        ushort_t* o = ab + (size_t)(i + 1) * INNER + h * DHEAD + 2 * lane;
        o[0] = f2bf(a0);
        o[1] = f2bf(a1);
    }
}

// ---------------------------------------------------------------------------
// Host-side launch
// ---------------------------------------------------------------------------
extern "C" void kernel_launch(void* const* d_in, const int* in_sizes, int n_in,
                              void* d_out, int out_size, void* d_ws, size_t ws_size,
                              hipStream_t stream) {
    (void)in_sizes; (void)n_in; (void)out_size; (void)ws_size;

    const float* x     = (const float*)d_in[0];  // 2048 x 512
    const float* w_qkv = (const float*)d_in[1];  // 512 x 1536
    const float* w_out = (const float*)d_in[2];  // 512 x 512
    const float* b_out = (const float*)d_in[3];  // 512
    float* out = (float*)d_out;                  // 2048 x 512 (f32)

    // Workspace carve-up (all region sizes multiples of 256B)
    char* p = (char*)d_ws;
    ushort_t* xb  = (ushort_t*)p; p += (size_t)MPAD * INNER * 2;   // bf16 x, padded rows zeroed
    ushort_t* wqb = (ushort_t*)p; p += (size_t)INNER * QKVW * 2;   // bf16 w_qkv
    ushort_t* wob = (ushort_t*)p; p += (size_t)INNER * INNER * 2;  // bf16 w_out
    float*    qkv = (float*)p;    p += (size_t)MPAD * QKVW * 4;    // f32 qkv (padded M)
    ushort_t* ab  = (ushort_t*)p;                                  // bf16 attn out (2048 x 512)

    // 1) Convert inputs to bf16 (with zero row padding for x)
    {
        int n_dst = MPAD * INNER, n_src = N_TOK * INNER;
        cvt_pad_kernel<<<(n_dst + 255) / 256, 256, 0, stream>>>(x, xb, n_src, n_dst);
    }
    {
        int n = INNER * QKVW;
        cvt_pad_kernel<<<(n + 255) / 256, 256, 0, stream>>>(w_qkv, wqb, n, n);
    }
    {
        int n = INNER * INNER;
        cvt_pad_kernel<<<(n + 255) / 256, 256, 0, stream>>>(w_out, wob, n, n);
    }

    // 2) QKV GEMM: qkv[MPAD x 1536] = xb[MPAD x 512] @ wqb[512 x 1536]
    {
        dim3 grid(MPAD / 16, QKVW / 64);
        gemm_bf16_kernel<<<grid, 128, 0, stream>>>(xb, wqb, qkv, nullptr,
                                                   INNER, QKVW, QKVW, INNER);
    }

    // 3) BOS output row (row 0 = v_bos)
    bos_row_kernel<<<2, 256, 0, stream>>>(qkv, ab);

    // 4) Windowed causal attention -> ab rows 1..2047 (bf16)
    attn_kernel<<<(HEADS * N_TOK) / 4, 128, 0, stream>>>(qkv, ab);

    // 5) Output projection: out[2048 x 512] = ab @ wob + b_out
    {
        dim3 grid(N_TOK / 16, INNER / 64);
        gemm_bf16_kernel<<<grid, 128, 0, stream>>>(ab, wob, out, b_out,
                                                   INNER, INNER, INNER, INNER);
    }
}
